// Sampler_65652870087160
// MI455X (gfx1250) — compile-verified, matrix operand order checked by reading, not run on hardware
//
#include <hip/hip_runtime.h>
#include <hip/hip_bf16.h>
#include <stdint.h>

#define SAMPLING_EPS 1e-5f
#define CH 4096   // streaming chunk: 16KB of floats per LDS buffer

typedef unsigned int u32;
typedef _Float16 f16;
typedef __attribute__((ext_vector_type(16))) _Float16 v16h;
typedef __attribute__((ext_vector_type(8)))  float    v8f;
typedef __attribute__((ext_vector_type(4)))  unsigned int u32x4;
typedef __attribute__((ext_vector_type(4)))  int      i32x4;
typedef __attribute__((ext_vector_type(8)))  int      i32x8;

#if __has_builtin(__builtin_amdgcn_tensor_load_to_lds) && __has_builtin(__builtin_amdgcn_s_wait_tensorcnt)
#define HAVE_TDM 1
#else
#define HAVE_TDM 0
#endif

// ---------------- small device helpers ----------------

__device__ __forceinline__ u32 fkey(float f) {
    // monotone float->uint key: larger float => larger key
    u32 u = __float_as_uint(f);
    return (u & 0x80000000u) ? ~u : (u | 0x80000000u);
}

__device__ __forceinline__ u32 hash10(u32 v) {
    return (v * 2654435761u) >> 22;   // 10-bit hash, 1024 slots
}

__device__ __forceinline__ void lse_merge(float& m, float& s, float m2, float s2) {
    if (s2 == 0.0f || m2 == -INFINITY) return;
    if (s == 0.0f || m == -INFINITY) { m = m2; s = s2; return; }
    if (m2 <= m) s += s2 * __expf(m2 - m);
    else { s = s2 + s * __expf(m - m2); m = m2; }
}

__device__ __forceinline__ float gumbel_noise(int b, int v) {
    u32 x = ((u32)b * 0x9E3779B1u) ^ ((u32)v * 0x85EBCA77u) ^ 0xA511E9B3u;
    x ^= x >> 16; x *= 0x7FEB352Du; x ^= x >> 15; x *= 0x846CA68Bu; x ^= x >> 16;
    float u = ((float)x + 0.5f) * (1.0f / 4294967296.0f);  // (0,1)
    return -__logf(-__logf(u));
}

__device__ __forceinline__ bool pair_gt(float va, int ia, float vb, int ib) {
    return (va > vb) || (va == vb && ia < ib);
}

// descending bitonic sort in LDS, n = power of two, blockDim >= n
__device__ void bitonic_desc(float* v, int* id, int n) {
    int tid = threadIdx.x;
    for (int k = 2; k <= n; k <<= 1) {
        for (int j = k >> 1; j > 0; j >>= 1) {
            __syncthreads();
            if (tid < n) {
                int ixj = tid ^ j;
                if (ixj > tid) {
                    float va = v[tid], vb = v[ixj];
                    int   ia = id[tid], ib = id[ixj];
                    bool up = ((tid & k) == 0);
                    bool sw = up ? pair_gt(vb, ib, va, ia) : pair_gt(va, ia, vb, ib);
                    if (sw) { v[tid] = vb; id[tid] = ib; v[ixj] = va; id[ixj] = ia; }
                }
            }
        }
    }
    __syncthreads();
}

// value with sparse penalty-patch applied (bitmap bit set -> hash lookup)
__device__ __forceinline__ float patched_val(float x, int v, const u32* bmLds,
                                             const u32* hkrow, const float* hvrow) {
    if ((bmLds[v >> 5] >> (v & 31)) & 1u) {
        u32 h = hash10((u32)v);
        while (hkrow[h] != (u32)v) h = (h + 1) & 1023u;
        x = hvrow[h];
    }
    return x;
}

// ---------------- TDM (Tensor Data Mover) 1D load -> LDS ----------------
// Minimal D# per CDNA5 ISA: 1D tile of `nelem` 4-byte
// elements, count=1, type=2, data_size=4B, tile_dim1=0 (unused).
__device__ __forceinline__ void tdm_load_1d(u32 lds_byte_off, const void* gptr, u32 nelem) {
#if HAVE_TDM
    unsigned long long ga = (unsigned long long)(uintptr_t)gptr;
    u32x4 g0;
    g0[0] = 1u;                                   // count=1, user mode
    g0[1] = lds_byte_off;                         // lds_addr
    g0[2] = (u32)ga;                              // global_addr[31:0]
    g0[3] = (u32)((ga >> 32) & 0x01FFFFFFu) | (2u << 30);  // global_addr[56:32] | type=2
    i32x8 g1;
    g1[0] = (int)(2u << 16);                      // data_size = 4 bytes
    g1[1] = (int)((nelem & 0xFFFFu) << 16);       // tensor_dim0[15:0]
    g1[2] = (int)((nelem >> 16) | (1u << 16));    // tensor_dim0[31:16], tensor_dim1 = 1
    g1[3] = (int)((nelem & 0xFFFFu) << 16);       // tile_dim0
    g1[4] = 0;                                    // tile_dim1 = 0 (unused), tile_dim2 = 0
    g1[5] = (int)nelem;                           // tensor_dim0_stride
    g1[6] = 0; g1[7] = 0;
    i32x4 z4 = {0, 0, 0, 0};
#if defined(__clang_major__) && (__clang_major__ >= 23)
    i32x8 z8 = {0, 0, 0, 0, 0, 0, 0, 0};
    __builtin_amdgcn_tensor_load_to_lds(g0, g1, z4, z4, z8, 0);
#else
    __builtin_amdgcn_tensor_load_to_lds(g0, g1, z4, z4, 0);
#endif
#else
    (void)lds_byte_off; (void)gptr; (void)nelem;
#endif
}

__device__ __forceinline__ void wait_tensor(int n) {
#if HAVE_TDM
    if (n == 0) __builtin_amdgcn_s_wait_tensorcnt((short)0);
    else        __builtin_amdgcn_s_wait_tensorcnt((short)1);
#else
    (void)n;
#endif
}

__device__ __forceinline__ u32 lds_off_of(const void* p) {
    // flat LDS aperture keeps the byte offset in addr[31:0]
    return (u32)(uintptr_t)p;
}

// ---------------- generic row streamer: TDM double-buffer (or direct B128) ----------------
// Calls process(v, x) for every element of the row. Must be invoked uniformly by the
// whole block (contains barriers). b0/b1 are 16B-aligned LDS buffers of CH floats.
template <typename Fn>
__device__ __forceinline__ void stream_row(const float* __restrict__ row, int V,
                                           float* b0, float* b1, Fn process) {
    int tid = threadIdx.x, bd = blockDim.x;
#if HAVE_TDM
    bool w0 = (tid >> 5) == 0;                    // wave 0 drives the TDM
    int nch = (V + CH - 1) / CH;
    if (w0) { int n0 = V < CH ? V : CH; tdm_load_1d(lds_off_of(b0), row, (u32)n0); }
    for (int c = 0; c < nch; ++c) {
        int nxt = c + 1;
        float* cur = (c & 1) ? b1 : b0;
        float* nbf = (c & 1) ? b0 : b1;
        if (w0 && nxt < nch) {
            int base = nxt * CH; int nn = V - base; if (nn > CH) nn = CH;
            tdm_load_1d(lds_off_of(nbf), row + base, (u32)nn);
        }
        if (w0) wait_tensor(nxt < nch ? 1 : 0);   // chunk c done (TENSORcnt is in-order)
        __syncthreads();
        int base = c * CH; int nn = V - base; if (nn > CH) nn = CH;
        int nn4 = nn >> 2;
        const float4* c4 = (const float4*)cur;    // ds_load_b128
        for (int j = tid; j < nn4; j += bd) {
            float4 x = c4[j];
            int v = base + (j << 2);
            process(v, x.x); process(v + 1, x.y); process(v + 2, x.z); process(v + 3, x.w);
        }
        for (int j = (nn4 << 2) + tid; j < nn; j += bd) process(base + j, cur[j]);
        __syncthreads();                          // buffer free before re-issue
    }
#else
    (void)b0; (void)b1;
    int V4 = V >> 2;
    const float4* r4 = (const float4*)row;        // global_load_b128
    for (int j = tid; j < V4; j += bd) {
        __builtin_prefetch((const float*)(r4 + j + bd), 0, 1);
        float4 x = r4[j];
        int v = j << 2;
        process(v, x.x); process(v + 1, x.y); process(v + 2, x.z); process(v + 3, x.w);
    }
    for (int v = (V4 << 2) + tid; v < V; v += bd) process(v, row[v]);
#endif
}

// ---------------- K0: workspace init ----------------
__global__ void init_kernel(u32* bitmap, size_t nbm, u32* hkey, size_t nhk) {
    size_t i = (size_t)blockIdx.x * blockDim.x + threadIdx.x;
    size_t st = (size_t)gridDim.x * blockDim.x;
    for (size_t j = i; j < nbm; j += st) bitmap[j] = 0u;
    for (size_t j = i; j < nhk; j += st) hkey[j] = 0xFFFFFFFFu;
}

// ---------------- K1: per-row log-sum-exp of raw logits (TDM streamed) ----------------
__global__ void lse_kernel(const float* __restrict__ logits, float* __restrict__ rowlse, int V) {
    __shared__ __attribute__((aligned(16))) float buf0[CH];
    __shared__ __attribute__((aligned(16))) float buf1[CH];
    __shared__ float wm[16], wsum[16];
    int b = blockIdx.x, tid = threadIdx.x, bd = blockDim.x;
    const float* row = logits + (size_t)b * V;
    float m = -INFINITY, s = 0.0f;

    stream_row(row, V, buf0, buf1, [&](int v, float x) {
        (void)v;
        lse_merge(m, s, x, 1.0f);
    });

    // wave32 butterfly reduce
    for (int off = 16; off; off >>= 1) {
        float m2 = __shfl_xor(m, off, 32), s2 = __shfl_xor(s, off, 32);
        lse_merge(m, s, m2, s2);
    }
    int wv = tid >> 5, ln = tid & 31;
    if (ln == 0) { wm[wv] = m; wsum[wv] = s; }
    __syncthreads();
    if (wv == 0) {
        int nw = bd >> 5;
        float m2 = (ln < nw) ? wm[ln] : -INFINITY;
        float s2 = (ln < nw) ? wsum[ln] : 0.0f;
        for (int off = 16; off; off >>= 1) {
            float m3 = __shfl_xor(m2, off, 32), s3 = __shfl_xor(s2, off, 32);
            lse_merge(m2, s2, m3, s3);
        }
        if (tid == 0) rowlse[b] = m2 + __logf(s2);
    }
}

// ---------------- K2: sparse penalties -> bitmap + hash of patched values ----------------
__global__ void penalty_kernel(const float* __restrict__ logits,
                               const int* __restrict__ prompt, const int* __restrict__ outtok,
                               const float* __restrict__ freq, const float* __restrict__ pres,
                               const float* __restrict__ rep,
                               u32* bitmap, u32* hkey, float* hval, int V, int P, int O) {
    __shared__ int toks[1024];
    int b = blockIdx.x, tid = threadIdx.x, bd = blockDim.x;
    int n = P + O; if (n > 1024) n = 1024;
    for (int i = tid; i < n; i += bd)
        toks[i] = (i < O) ? outtok[(size_t)b * O + i] : prompt[(size_t)b * P + (i - O)];
    __syncthreads();
    u32*   bmrow = bitmap + (size_t)b * 4096;
    u32*   hkrow = hkey   + (size_t)b * 1024;
    float* hvrow = hval   + (size_t)b * 1024;
    float fp = freq[b], pp = pres[b], rp = rep[b];
    for (int i = tid; i < n; i += bd) {
        int t = toks[i];
        bool first = true;
        for (int j = 0; j < i; ++j) if (toks[j] == t) { first = false; break; }
        if (!first) continue;                    // exactly one writer per unique token
        int cnt = 0;
        int olim = O < n ? O : n;
        for (int j = 0; j < olim; ++j) cnt += (toks[j] == t);
        float xv = logits[(size_t)b * V + t];
        xv = (xv > 0.0f) ? xv / rp : xv * rp;    // repetition penalty (seen == true here)
        if (cnt > 0) xv -= fp * (float)cnt + pp; // frequency + presence
        u32 h = hash10((u32)t);
        for (;;) {
            u32 old = atomicCAS(&hkrow[h], 0xFFFFFFFFu, (u32)t);
            if (old == 0xFFFFFFFFu) break;
            h = (h + 1) & 1023u;
        }
        hvrow[h] = xv;
        atomicOr(&bmrow[t >> 5], 1u << (t & 31));
    }
}

// ---------------- K3/K4: 2-pass radix select (64K-bin LDS histogram, TDM-fed) --------
// Finds the key of the k-th largest element; 256KB LDS histogram => only 2 passes.
__global__ void select_kernel(const float* __restrict__ logits,
                              const u32* __restrict__ bitmap, const u32* __restrict__ hkey,
                              const float* __restrict__ hval,
                              int usePatch, int V, int kFixed, const int* __restrict__ dnlp,
                              u32* __restrict__ outkey) {
    extern __shared__ u32 smem[];
    u32*   hist = smem;                                  // 65536 bins (256KB)
    u32*   bm   = smem + 65536;                          // 4096 words if usePatch
    float* st0  = (float*)(smem + 65536 + (usePatch ? 4096 : 0));  // 16B aligned
    float* st1  = st0 + CH;
    __shared__ u32 coarse[512];
    __shared__ int s_prefix, s_above;
    int b = blockIdx.x, tid = threadIdx.x, bd = blockDim.x;
    const float* row   = logits + (size_t)b * V;
    const u32*   bmrow = bitmap + (size_t)b * 4096;
    const u32*   hkrow = hkey   + (size_t)b * 1024;
    const float* hvrow = hval   + (size_t)b * 1024;
    int k = (kFixed > 0) ? kFixed : *dnlp;
    if (k < 1) k = 1; if (k > 128) k = 128;

    for (int i = tid; i < 65536; i += bd) hist[i] = 0u;
    if (usePatch) for (int i = tid; i < 4096; i += bd) bm[i] = bmrow[i];
    __syncthreads();

    // pass A: histogram of high 16 key bits
    stream_row(row, V, st0, st1, [&](int v, float x) {
        if (usePatch) x = patched_val(x, v, bm, hkrow, hvrow);
        atomicAdd(&hist[fkey(x) >> 16], 1u);
    });
    __syncthreads();
    if (tid < 512) {                  // descending coarse sums, 128 bins each
        u32 s = 0; int hi = 65536 - tid * 128;
        for (int i = hi - 128; i < hi; ++i) s += hist[i];
        coarse[tid] = s;
    }
    __syncthreads();
    if (tid == 0) {
        u32 run = 0; int pfx = 0; u32 above = 0;
        for (int t = 0; t < 512; ++t) {
            if (run + coarse[t] >= (u32)k) {
                int hi = 65536 - t * 128;
                for (int i = hi - 1; i >= hi - 128; --i) {
                    if (run + hist[i] >= (u32)k) { pfx = i; above = run; break; }
                    run += hist[i];
                }
                break;
            }
            run += coarse[t];
        }
        s_prefix = pfx; s_above = (int)above;
    }
    __syncthreads();
    int prefix = s_prefix, above = s_above;
    for (int i = tid; i < 65536; i += bd) hist[i] = 0u;
    __syncthreads();

    // pass B: low 16 bits among elements in the prefix bin
    stream_row(row, V, st0, st1, [&](int v, float x) {
        if (usePatch) x = patched_val(x, v, bm, hkrow, hvrow);
        u32 key = fkey(x);
        if ((int)(key >> 16) == prefix) atomicAdd(&hist[key & 0xFFFFu], 1u);
    });
    __syncthreads();
    if (tid < 512) {
        u32 s = 0; int hi = 65536 - tid * 128;
        for (int i = hi - 128; i < hi; ++i) s += hist[i];
        coarse[tid] = s;
    }
    __syncthreads();
    if (tid == 0) {
        int kk = k - above; u32 run = 0; int low = 0;
        for (int t = 0; t < 512; ++t) {
            if (run + coarse[t] >= (u32)kk) {
                int hi = 65536 - t * 128;
                for (int i = hi - 1; i >= hi - 128; --i) {
                    if (run + hist[i] >= (u32)kk) { low = i; break; }
                    run += hist[i];
                }
                break;
            }
            run += coarse[t];
        }
        outkey[b] = ((u32)prefix << 16) | (u32)low;
    }
}

// ---------------- WMMA prefix-sum: cum = LowerTriOnes(64x64) * p, two 64-chunks ----------
// A built per documented 16-bit A 16x32 layout; B is p replicated across all 16 columns.
__device__ void wmma_cumsum128(const float* __restrict__ p, float* __restrict__ cum) {
    int lane = threadIdx.x;           // caller guarantees tid < 32, full wave active
    int laneM = lane & 15;
    int hiK = lane >> 4;
    for (int c = 0; c < 2; ++c) {
        int base = c * 64;
        for (int i = 0; i < 4; ++i) {
            v8f acc = {0, 0, 0, 0, 0, 0, 0, 0};
            for (int kb = 0; kb < 2; ++kb) {
                v16h a, bv;
#pragma unroll
                for (int t = 0; t < 16; ++t) {
                    int K = (t < 8) ? (t + 8 * hiK) : (t + 8 + 8 * hiK);
                    int col = kb * 32 + K;
                    int rowi = i * 16 + laneM;
                    a[t]  = (f16)((col <= rowi) ? 1.0f : 0.0f);
                    bv[t] = (f16)p[base + col];
                }
                acc = __builtin_amdgcn_wmma_f32_16x16x32_f16(
                    false, a, false, bv, (short)0, acc, false, false);
            }
            if (laneM == 0) {
#pragma unroll
                for (int r = 0; r < 8; ++r)
                    cum[base + i * 16 + r + 8 * hiK] = acc[r];  // D: VGPR r -> M=r (+8 hi half)
            }
        }
    }
}

// ---------------- K5: gather top candidates, top-k/top-p threshold, sample ----------------
__global__ void sample_kernel(const float* __restrict__ logits,
                              const u32* __restrict__ bitmap, const u32* __restrict__ hkey,
                              const float* __restrict__ hval,
                              const float* __restrict__ temperature, const float* __restrict__ top_p,
                              const int* __restrict__ top_k, const u32* __restrict__ t64key,
                              int V, int* __restrict__ samp_ws, int* __restrict__ out_i) {
    __shared__ u32 bm[4096];
    __shared__ float vals[128]; __shared__ int idxs[128];
    __shared__ float pbuf[128]; __shared__ float cum[128];
    __shared__ int cnt; __shared__ float red[4];
    int b = blockIdx.x, tid = threadIdx.x, bd = blockDim.x;
    const float* row   = logits + (size_t)b * V;
    const u32*   bmrow = bitmap + (size_t)b * 4096;
    const u32*   hkrow = hkey   + (size_t)b * 1024;
    const float* hvrow = hval   + (size_t)b * 1024;
    for (int i = tid; i < 4096; i += bd) bm[i] = bmrow[i];
    if (tid == 0) cnt = 0;
    __syncthreads();
    u32 tkey = t64key[b];
    auto gather = [&](int v, float x) {
        x = patched_val(x, v, bm, hkrow, hvrow);
        if (fkey(x) >= tkey) {
            int p = atomicAdd(&cnt, 1);
            if (p < 128) { vals[p] = x; idxs[p] = v; }
        }
    };
    {
        int V4 = V >> 2;
        const float4* r4 = (const float4*)row;    // global_load_b128
        for (int q = tid; q < V4; q += bd) {
            __builtin_prefetch((const float*)(r4 + q + bd), 0, 1);
            float4 x4 = r4[q];
            int v = q << 2;
            gather(v, x4.x); gather(v + 1, x4.y); gather(v + 2, x4.z); gather(v + 3, x4.w);
        }
        for (int v = (V4 << 2) + tid; v < V; v += bd) gather(v, row[v]);
    }
    __syncthreads();
    int n = cnt; if (n > 128) n = 128;
    for (int i = tid; i < 128; i += bd) if (i >= n) { vals[i] = -INFINITY; idxs[i] = 0x7FFFFFFF; }
    bitonic_desc(vals, idxs, 128);     // desc by value, asc by index on ties

    float temp = temperature[b];
    float ts = (temp < SAMPLING_EPS) ? 1.0f : temp;
    int kq = top_k[b]; if (kq < 1) kq = 1; if (kq > 128) kq = 128;
    float kthv = vals[kq - 1];
    float m0 = vals[0] / ts;
    if (tid < 128) {
        bool kept = (vals[tid] >= kthv) && (vals[tid] > -INFINITY);
        pbuf[tid] = kept ? __expf(vals[tid] / ts - m0) : 0.0f;
    }
    __syncthreads();
    if (tid == 0) { float s = 0; for (int i = 0; i < 128; ++i) s += pbuf[i]; red[0] = s; }
    __syncthreads();
    if (tid < 128) pbuf[tid] /= red[0];
    __syncthreads();
    if (tid < 32) wmma_cumsum128(pbuf, cum);   // wave 0, EXEC all ones
    __syncthreads();
    if (tid < 64) cum[64 + tid] += cum[63];    // chain the two 64-chunks
    __syncthreads();
    float tp = top_p[b];
    if (tid == 0) {
        int nk = 0;
        for (int i = 0; i < 128; ++i) {
            bool kept = (vals[i] >= kthv) && (vals[i] > -INFINITY);
            if (kept && (cum[i] - pbuf[i] < tp)) nk++;
        }
        if (nk < 1) nk = 1;
        red[1] = fmaxf(kthv / ts, vals[nk - 1] / ts);   // thresh = max(kth, pth)
    }
    __syncthreads();
    float thresh = red[1];
    if (tid == 0) {
        float best = -INFINITY; int bi = 0x7FFFFFFF;
        for (int i = 0; i < 128; ++i) {
            if (vals[i] > -INFINITY) {
                float s = vals[i] / ts;
                if (s >= thresh) {
                    float sc = s + gumbel_noise(b, idxs[i]);
                    if (sc > best || (sc == best && idxs[i] < bi)) { best = sc; bi = idxs[i]; }
                }
            }
        }
        int greedy = idxs[0];
        int sampled = (temp < SAMPLING_EPS) ? greedy : bi;
        samp_ws[b] = sampled;
        out_i[b] = sampled;
    }
}

// ---------------- K6: top-num_logprobs of raw log-softmax, token logprob, ranks ----------
__global__ void logprob_kernel(const float* __restrict__ logits, const float* __restrict__ rowlse,
                               const u32* __restrict__ tnkey, const int* __restrict__ samp_ws,
                               const int* __restrict__ dnlp, int V, int B, void* dout) {
    __shared__ float gv[64]; __shared__ int gi[64];
    __shared__ int cnt; __shared__ int rnk;
    int b = blockIdx.x, tid = threadIdx.x, bd = blockDim.x;
    if (tid == 0) { cnt = 0; rnk = 0; }
    __syncthreads();
    const float* row = logits + (size_t)b * V;
    int sampled = samp_ws[b];
    float sl = row[sampled];
    u32 tk = tnkey[b];
    int local = 0;
    auto visit = [&](int v, float lv) {
        if (fkey(lv) >= tk) {
            int p = atomicAdd(&cnt, 1);
            if (p < 64) { gv[p] = lv; gi[p] = v; }
        }
        local += (lv >= sl) ? 1 : 0;
    };
    {
        int V4 = V >> 2;
        const float4* r4 = (const float4*)row;    // global_load_b128
        for (int q = tid; q < V4; q += bd) {
            __builtin_prefetch((const float*)(r4 + q + bd), 0, 1);
            float4 x4 = r4[q];
            int v = q << 2;
            visit(v, x4.x); visit(v + 1, x4.y); visit(v + 2, x4.z); visit(v + 3, x4.w);
        }
        for (int v = (V4 << 2) + tid; v < V; v += bd) visit(v, row[v]);
    }
    for (int off = 16; off; off >>= 1) local += __shfl_xor(local, off, 32);
    if ((tid & 31) == 0) atomicAdd(&rnk, local);
    __syncthreads();
    int n = cnt; if (n > 64) n = 64;
    for (int i = tid; i < 64; i += bd) if (i >= n) { gv[i] = -INFINITY; gi[i] = 0x7FFFFFFF; }
    bitonic_desc(gv, gi, 64);
    int nlp = *dnlp; if (nlp < 0) nlp = 0; if (nlp > 64) nlp = 64;
    float lse = rowlse[b];
    int*   oi = (int*)dout;
    float* of = (float*)dout;
    int idxBase = B + b * (nlp + 1);
    int lpBase  = B + B * (nlp + 1) + b * (nlp + 1);
    for (int j = tid; j < nlp; j += bd) {
        oi[idxBase + j] = gi[j];
        of[lpBase + j]  = gv[j] - lse;
    }
    if (tid == 0) {
        oi[idxBase + nlp] = sampled;
        of[lpBase + nlp]  = sl - lse;
        oi[B + 2 * B * (nlp + 1) + b] = rnk;
    }
}

// ---------------- host-side orchestration ----------------
extern "C" void kernel_launch(void* const* d_in, const int* in_sizes, int n_in,
                              void* d_out, int out_size, void* d_ws, size_t ws_size,
                              hipStream_t stream) {
    const float* logits      = (const float*)d_in[0];
    const float* temperature = (const float*)d_in[1];
    const float* top_p       = (const float*)d_in[2];
    const float* freq        = (const float*)d_in[3];
    const float* pres        = (const float*)d_in[4];
    const float* rep         = (const float*)d_in[5];
    const int*   topk        = (const int*)d_in[6];
    const int*   prompt      = (const int*)d_in[7];
    const int*   outtok      = (const int*)d_in[8];
    const int*   dnlp        = (const int*)d_in[9];
    int B = in_sizes[1];
    int V = in_sizes[0] / B;
    int P = in_sizes[7] / B;
    int O = in_sizes[8] / B;

    char* w = (char*)d_ws;
    auto carve = [&](size_t bytes) { char* p = w; w += (bytes + 255) & ~(size_t)255; return p; };
    u32*   bitmap = (u32*)carve((size_t)B * 4096 * 4);   // 16KB/row bitmap (V<=131072)
    u32*   hkey   = (u32*)carve((size_t)B * 1024 * 4);
    float* hval   = (float*)carve((size_t)B * 1024 * 4);
    float* rowlse = (float*)carve((size_t)B * 4);
    u32*   t64    = (u32*)carve((size_t)B * 4);
    u32*   tnk    = (u32*)carve((size_t)B * 4);
    int*   samp   = (int*)carve((size_t)B * 4);

    init_kernel<<<256, 256, 0, stream>>>(bitmap, (size_t)B * 4096, hkey, (size_t)B * 1024);
    lse_kernel<<<B, 512, 0, stream>>>(logits, rowlse, V);

    int pbd = P + O; if (pbd > 1024) pbd = 1024; pbd = ((pbd + 63) / 64) * 64;
    penalty_kernel<<<B, pbd, 0, stream>>>(logits, prompt, outtok, freq, pres, rep,
                                          bitmap, hkey, hval, V, P, O);

    // hist(256KB) [+ bitmap(16KB)] + 2x16KB TDM stage buffers, within 320KB/WGP
    size_t shPatch = (size_t)(65536 + 4096 + 2 * CH) * 4;
    size_t shPlain = (size_t)(65536 + 2 * CH) * 4;
    select_kernel<<<B, 512, shPatch, stream>>>(logits, bitmap, hkey, hval, 1, V, 64, dnlp, t64);
    select_kernel<<<B, 512, shPlain, stream>>>(logits, bitmap, hkey, hval, 0, V, 0,  dnlp, tnk);

    sample_kernel<<<B, 256, 0, stream>>>(logits, bitmap, hkey, hval,
                                         temperature, top_p, topk, t64, V, samp, (int*)d_out);
    logprob_kernel<<<B, 512, 0, stream>>>(logits, rowlse, tnk, samp, dnlp, V, B, d_out);
}
